// PredictLayer_42606075576599
// MI455X (gfx1250) — matrix operand order repaired and checked by reading, not compiled, not run on hardware
//
#include <hip/hip_runtime.h>
#include <hip/hip_bf16.h>

typedef __bf16 bf16_t;
typedef __attribute__((ext_vector_type(16))) __bf16 bf16x16;
typedef __attribute__((ext_vector_type(8)))  float  floatx8;

#define DEV __device__ __forceinline__

// ---------------------------------------------------------------------------
// Packed fragment layout (CDNA5 WMMA 16x16x32 bf16, wave32) — see ISA 7.12.2.
// A tile: lane = lane_hi*16 + (m&15), lane_hi = (k&31)>>4 & 1,
//         slot = k&15.
// Buffers are tile-major: [(mt*KT + kt)*32 + lane]*16 + slot, so the GEMM
// inner loop streams 32 contiguous bytes per lane per fragment.
// ---------------------------------------------------------------------------

DEV void pack_store(bf16_t* P, int KT, int m, int k, bf16_t v) {
  int mt = m >> 4, ml = m & 15, kt = k >> 5, kk = k & 31;
  int lh = (kk >> 4) & 1;
  int slot = kk & 15;
  P[(((size_t)mt * KT + kt) * 32 + (size_t)(lh * 16 + ml)) * 16 + slot] = v;
}

DEV float sigf(float x) { return 1.0f / (1.0f + expf(-x)); }

// ---------------------------------------------------------------------------
// GEMM: C[M x N] = Apack[M x K] * Bpack[K x N] + bias, optional ReLU.
// One wave -> 16x64 strip (4 WMMA tiles sharing one A fragment).
// B fragments stream Global->LDS via async copies (ASYNCcnt), double-buffered;
// k-loop is unrolled x2 so each half uses a compile-time stage offset and its
// own A registers (no rotation moves, no WMMA->VALU hazard NOPs).
// KT must be even (96/64/32/8 here).
// ---------------------------------------------------------------------------
__global__ __launch_bounds__(256) void gemm_bf16_wmma(
    const bf16_t* __restrict__ Ap, const bf16_t* __restrict__ Bp,
    const float* __restrict__ bias, int nbias,
    float* __restrict__ Cf, bf16_t* __restrict__ Cpack, int relu,
    int N, int KT, int totalWaves)
{
  // 8 waves * 2 stages * 4 fragments * 1KB
  __shared__ unsigned char smem[65536] __attribute__((aligned(128)));

  int w = (int)((blockIdx.x * blockDim.x + threadIdx.x) >> 5);
  int lane = threadIdx.x & 31;
  if (w >= totalWaves) return;            // uniform per wave: EXEC stays all-1s
  int n4 = N >> 6;
  int mt = w / n4, n4i = w % n4;
  unsigned wbase = (unsigned)(threadIdx.x >> 5) * 8192u;

  const bf16_t* arow  = Ap + (size_t)mt * KT * 512 + (size_t)lane * 16;
  const bf16_t* bbase = Bp + (size_t)(n4i * 4) * KT * 512 + (size_t)lane * 16;

  // one double-buffer stage of async Global->LDS copies: 8 x 16B per lane
  auto issue_stage = [&](int kt, unsigned stageOff) {
#pragma unroll
    for (int j = 0; j < 4; ++j) {
      const bf16_t* g = bbase + ((size_t)j * KT + kt) * 512;
      unsigned l = (unsigned)(uintptr_t)(smem + wbase + stageOff +
                                         (unsigned)j * 1024u +
                                         (unsigned)lane * 32u);
      // INST_OFFSET applies to both LDS and global address (ISA async ldst)
      asm volatile(
          "global_load_async_to_lds_b128 %0, %1, off\n\t"
          "global_load_async_to_lds_b128 %0, %1, off offset:16"
          :: "v"(l), "v"(g) : "memory");
    }
  };

  floatx8 acc[4] = {};

  // consume one resident stage: clause 8 ds_loads, then 4 back-to-back WMMAs
  auto compute = [&](bf16x16 a, unsigned stageOff) {
    const bf16x16* bfr =
        (const bf16x16*)(smem + wbase + stageOff + (unsigned)lane * 32u);
    bf16x16 b0 = bfr[0];
    bf16x16 b1 = bfr[32];
    bf16x16 b2 = bfr[64];
    bf16x16 b3 = bfr[96];
    acc[0] = __builtin_amdgcn_wmma_f32_16x16x32_bf16(false, a, false, b0,
                                                     (short)0, acc[0], false, false);
    acc[1] = __builtin_amdgcn_wmma_f32_16x16x32_bf16(false, a, false, b1,
                                                     (short)0, acc[1], false, false);
    acc[2] = __builtin_amdgcn_wmma_f32_16x16x32_bf16(false, a, false, b2,
                                                     (short)0, acc[2], false, false);
    acc[3] = __builtin_amdgcn_wmma_f32_16x16x32_bf16(false, a, false, b3,
                                                     (short)0, acc[3], false, false);
  };

  issue_stage(0, 0u);
  bf16x16 a0 = *(const bf16x16*)(arow);

  for (int kt = 0; kt < KT; kt += 2) {
    // stage kt resident after this wait; stage kt+1 in flight
    issue_stage(kt + 1, 4096u);
    asm volatile("s_wait_asynccnt 0x8" ::: "memory");
    bf16x16 a1 = *(const bf16x16*)(arow + (size_t)(kt + 1) * 512);
    compute(a0, 0u);

    if (kt + 2 < KT) {
      issue_stage(kt + 2, 0u);
      asm volatile("s_wait_asynccnt 0x8" ::: "memory");  // kt+1 resident
      a0 = *(const bf16x16*)(arow + (size_t)(kt + 2) * 512);
    } else {
      asm volatile("s_wait_asynccnt 0x0" ::: "memory");
    }
    compute(a1, 4096u);
  }

  int rbase = mt * 16 + ((lane >> 4) << 3);  // C/D layout: lanes 16-31 -> rows +8
  int cbase = n4i * 64 + (lane & 15);
#pragma unroll
  for (int j = 0; j < 4; ++j) {
    int c = cbase + j * 16;
    float badd = (bias && c < nbias) ? bias[c] : 0.0f;
#pragma unroll
    for (int r = 0; r < 8; ++r) {
      float v = acc[j][r] + badd;
      if (relu) v = fmaxf(v, 0.0f);
      if (Cf)    Cf[(size_t)(rbase + r) * N + c] = v;
      if (Cpack) pack_store(Cpack, N >> 5, rbase + r, c, (bf16_t)v);
    }
  }
}

// ---------------------------------------------------------------------------
// Weight pack: Bpack[k][n] = k<K1 ? W1[n][k] : W2[n][k-K1]   (transposed cat)
// ---------------------------------------------------------------------------
__global__ void pack_weightT_cat(const float* __restrict__ W1,
                                 const float* __restrict__ W2,
                                 int K1, int K, int N, bf16_t* __restrict__ Bp)
{
  int tid = blockIdx.x * blockDim.x + threadIdx.x;
  int KT = K >> 5;
  int total = (N >> 4) * KT * 32;
  if (tid >= total) return;
  int lane = tid & 31;
  int kt = (tid >> 5) % KT;
  int nt = (tid >> 5) / KT;
  int n  = nt * 16 + (lane & 15);
  int k0 = kt * 32 + ((lane >> 4) << 4);
  const float* src = (k0 < K1) ? (W1 + (size_t)n * K1 + k0)
                               : (W2 + (size_t)n * (K - K1) + (k0 - K1));
  bf16_t* dst = Bp + (((size_t)nt * KT + kt) * 32 + lane) * 16;
#pragma unroll
  for (int s = 0; s < 16; ++s) dst[s] = (bf16_t)src[s];
}

// Bpack[k][n] = W[k][n] (no transpose), zero-padded to K x N.
__global__ void pack_b_notrans(const float* __restrict__ W, int Ksrc, int Nsrc,
                               int K, int N, bf16_t* __restrict__ Bp)
{
  int tid = blockIdx.x * blockDim.x + threadIdx.x;
  int KT = K >> 5;
  int total = (N >> 4) * KT * 32;
  if (tid >= total) return;
  int lane = tid & 31;
  int kt = (tid >> 5) % KT;
  int nt = (tid >> 5) / KT;
  int n  = nt * 16 + (lane & 15);
  int k0 = kt * 32 + ((lane >> 4) << 4);
  bf16_t* dst = Bp + (((size_t)nt * KT + kt) * 32 + lane) * 16;
#pragma unroll
  for (int s = 0; s < 16; ++s) {
    int k = k0 + s;
    float v = (k < Ksrc && n < Nsrc) ? W[(size_t)k * Nsrc + n] : 0.0f;
    dst[s] = (bf16_t)v;
  }
}

// Pack joint = concat(data1, data2) into Abig rows m = t*64 + b (time-major),
// K-columns 0..2047 of the 3072-wide packed A (h1 occupies 2048..3071).
__global__ void pack_joint(const float* __restrict__ d1,
                           const float* __restrict__ d2,
                           bf16_t* __restrict__ Abig)
{
  const int T = 128, KTfull = 96, KTj = 64;
  int tid = blockIdx.x * blockDim.x + threadIdx.x;
  int total = ((T * 64) >> 4) * KTj * 32;  // 1,048,576
  if (tid >= total) return;
  int lane = tid & 31;
  int kt = (tid >> 5) % KTj;
  int mt = (tid >> 5) / KTj;
  int m  = mt * 16 + (lane & 15);
  int t  = m >> 6, b = m & 63;
  int k0 = kt * 32 + ((lane >> 4) << 4);
  const float* src = (k0 < 1024)
      ? d1 + ((size_t)b * T + t) * 1024 + k0
      : d2 + ((size_t)b * T + t) * 1024 + (k0 - 1024);
  bf16_t* dst = Abig + (((size_t)mt * KTfull + kt) * 32 + lane) * 16;
#pragma unroll
  for (int s = 0; s < 16; ++s) dst[s] = (bf16_t)src[s];
}

// ---------------------------------------------------------------------------
// Layer-1 cell: gates [64 x 4096] (i,f,g,o) -> c1, h1 fp32, h1 packed into the
// layer-2 A buffer (cols 0..1023) and the NEXT step's Abig (cols 2048..3071).
// ---------------------------------------------------------------------------
__global__ void lstm_cell1(const float* __restrict__ gates,
                           float* __restrict__ c, float* __restrict__ hf,
                           bf16_t* __restrict__ AbigNext,
                           bf16_t* __restrict__ Acat)
{
  int idx = blockIdx.x * blockDim.x + threadIdx.x;
  if (idx >= 64 * 1024) return;
  int b = idx >> 10, j = idx & 1023;
  const float* g = gates + (size_t)b * 4096;
  float gi = g[j], gf = g[1024 + j], gg = g[2048 + j], go = g[3072 + j];
  float cn = sigf(gf) * c[idx] + sigf(gi) * tanhf(gg);
  c[idx] = cn;
  float h = sigf(go) * tanhf(cn);
  hf[idx] = h;
  bf16_t hb = (bf16_t)h;
  pack_store(Acat, 64, b, j, hb);                           // K=2048 (KT=64)
  if (AbigNext) pack_store(AbigNext, 96, b, 2048 + j, hb);  // K=3072 (KT=96)
}

// Layer-2 cell: -> c2, h2 packed into Acat cols 1024..2047 (next step), and
// out_t = h1 + h2 packed bf16 into outpack rows t*64+b (K=1024).
__global__ void lstm_cell2(const float* __restrict__ gates,
                           float* __restrict__ c, const float* __restrict__ h1f,
                           bf16_t* __restrict__ Acat,
                           bf16_t* __restrict__ outpack, int t)
{
  int idx = blockIdx.x * blockDim.x + threadIdx.x;
  if (idx >= 64 * 1024) return;
  int b = idx >> 10, j = idx & 1023;
  const float* g = gates + (size_t)b * 4096;
  float gi = g[j], gf = g[1024 + j], gg = g[2048 + j], go = g[3072 + j];
  float cn = sigf(gf) * c[idx] + sigf(gi) * tanhf(gg);
  c[idx] = cn;
  float h2 = sigf(go) * tanhf(cn);
  pack_store(Acat, 64, b, 1024 + j, (bf16_t)h2);
  float o = h1f[idx] + h2;
  pack_store(outpack, 32, t * 64 + b, j, (bf16_t)o);        // K=1024 (KT=32)
}

// One wave32 per row; shuffle reductions. logits [8192 x 1024], row m=t*64+b;
// output goes to [b][t][0..999].
__global__ void softmax_rows(const float* __restrict__ logits,
                             float* __restrict__ out)
{
  int w = (int)((blockIdx.x * blockDim.x + threadIdx.x) >> 5);
  int lane = threadIdx.x & 31;
  if (w >= 8192) return;
  const float* row = logits + (size_t)w * 1024;
  int b = w & 63, t = w >> 6;
  float* orow = out + ((size_t)b * 128 + t) * 1000;
  float mx = -3.0e38f;
  for (int c = lane; c < 1000; c += 32) mx = fmaxf(mx, row[c]);
#pragma unroll
  for (int off = 16; off > 0; off >>= 1) mx = fmaxf(mx, __shfl_xor(mx, off, 32));
  float s = 0.0f;
  for (int c = lane; c < 1000; c += 32) s += expf(row[c] - mx);
#pragma unroll
  for (int off = 16; off > 0; off >>= 1) s += __shfl_xor(s, off, 32);
  float inv = 1.0f / s;
  for (int c = lane; c < 1000; c += 32) orow[c] = expf(row[c] - mx) * inv;
}

// ---------------------------------------------------------------------------
extern "C" void kernel_launch(void* const* d_in, const int* in_sizes, int n_in,
                              void* d_out, int out_size, void* d_ws, size_t ws_size,
                              hipStream_t stream)
{
  const float* data1 = (const float*)d_in[0];
  const float* data2 = (const float*)d_in[1];
  const float* W_ih1 = (const float*)d_in[2];
  const float* W_hh1 = (const float*)d_in[3];
  const float* b1    = (const float*)d_in[4];
  const float* W_ih2 = (const float*)d_in[5];
  const float* W_hh2 = (const float*)d_in[6];
  const float* b2    = (const float*)d_in[7];
  const float* Wp1   = (const float*)d_in[8];
  const float* bp1   = (const float*)d_in[9];
  const float* Wp2   = (const float*)d_in[10];
  const float* bp2   = (const float*)d_in[11];
  float* out = (float*)d_out;

  char* ws = (char*)d_ws;
  size_t off = 0;
  auto alloc = [&](size_t bytes) -> char* {
    char* p = ws + off;
    off += (bytes + 255) & ~(size_t)255;
    return p;
  };
  bf16_t* Bw1     = (bf16_t*)alloc((size_t)4096 * 3072 * 2);  // [Wih1;Whh1]^T
  bf16_t* Bw2     = (bf16_t*)alloc((size_t)4096 * 2048 * 2);  // [Wih2;Whh2]^T
  bf16_t* Bp1     = (bf16_t*)alloc((size_t)1024 * 256 * 2);
  bf16_t* Bp2     = (bf16_t*)alloc((size_t)256 * 1024 * 2);   // N pad 1000->1024
  bf16_t* Abig    = (bf16_t*)alloc((size_t)8192 * 3072 * 2);  // [x_t | h1] packed
  bf16_t* Acat    = (bf16_t*)alloc((size_t)64 * 2048 * 2);    // [h1 | h2] packed
  float*  gates1  = (float*)alloc((size_t)64 * 4096 * 4);
  float*  gates2  = (float*)alloc((size_t)64 * 4096 * 4);
  float*  c1      = (float*)alloc((size_t)64 * 1024 * 4);
  float*  c2      = (float*)alloc((size_t)64 * 1024 * 4);
  float*  h1f     = (float*)alloc((size_t)64 * 1024 * 4);
  bf16_t* outpack = (bf16_t*)alloc((size_t)8192 * 1024 * 2);
  bf16_t* hidpack = (bf16_t*)alloc((size_t)8192 * 256 * 2);
  float*  logits  = (float*)alloc((size_t)8192 * 1024 * 4);
  (void)ws_size; (void)in_sizes; (void)n_in; (void)out_size;

  // zero initial state: h-columns of Abig row-block 0, Acat, c1, c2
  hipMemsetAsync(Abig, 0, (size_t)4 * 96 * 512 * 2, stream);
  hipMemsetAsync(Acat, 0, (size_t)64 * 2048 * 2, stream);
  hipMemsetAsync(c1, 0, (size_t)64 * 1024 * 4, stream);
  hipMemsetAsync(c2, 0, (size_t)64 * 1024 * 4, stream);

  // pack weights + inputs (bf16, WMMA fragment layout)
  { int total = (4096 >> 4) * (3072 >> 5) * 32;
    pack_weightT_cat<<<(total + 255) / 256, 256, 0, stream>>>(W_ih1, W_hh1, 2048, 3072, 4096, Bw1); }
  { int total = (4096 >> 4) * (2048 >> 5) * 32;
    pack_weightT_cat<<<(total + 255) / 256, 256, 0, stream>>>(W_ih2, W_hh2, 1024, 2048, 4096, Bw2); }
  { int total = (256 >> 4) * (1024 >> 5) * 32;
    pack_b_notrans<<<(total + 255) / 256, 256, 0, stream>>>(Wp1, 1024, 256, 1024, 256, Bp1); }
  { int total = (1024 >> 4) * (256 >> 5) * 32;
    pack_b_notrans<<<(total + 255) / 256, 256, 0, stream>>>(Wp2, 256, 1000, 256, 1024, Bp2); }
  { int total = (8192 >> 4) * 64 * 32;
    pack_joint<<<(total + 255) / 256, 256, 0, stream>>>(data1, data2, Abig); }

  // recurrent scan: per step two WMMA GEMMs + two cell kernels
  const int wavesStep = 4 * (4096 / 64);        // 256 waves, 32 blocks
  const int blocksStep = (wavesStep * 32) / 256;
  for (int t = 0; t < 128; ++t) {
    const bf16_t* At = Abig + (size_t)(t * 4) * 96 * 512;
    gemm_bf16_wmma<<<blocksStep, 256, 0, stream>>>(
        At, Bw1, b1, 4096, gates1, nullptr, 0, 4096, 96, wavesStep);
    bf16_t* AbigNext = (t < 127) ? (Abig + (size_t)((t + 1) * 4) * 96 * 512) : nullptr;
    lstm_cell1<<<256, 256, 0, stream>>>(gates1, c1, h1f, AbigNext, Acat);
    gemm_bf16_wmma<<<blocksStep, 256, 0, stream>>>(
        Acat, Bw2, b2, 4096, gates2, nullptr, 0, 4096, 64, wavesStep);
    lstm_cell2<<<256, 256, 0, stream>>>(gates2, c2, h1f, Acat, outpack, t);
  }

  // MLP head: hid = relu(out @ Wp1 + bp1) stored directly as packed bf16 A
  { int waves = (8192 / 16) * (256 / 64);       // 2048
    gemm_bf16_wmma<<<(waves * 32) / 256, 256, 0, stream>>>(
        outpack, Bp1, bp1, 256, nullptr, hidpack, 1, 256, 32, waves); }
  // logits = hid @ Wp2 + bp2 (N padded to 1024; pad cols have zero weights)
  { int waves = (8192 / 16) * (1024 / 64);      // 8192
    gemm_bf16_wmma<<<(waves * 32) / 256, 256, 0, stream>>>(
        hidpack, Bp2, bp2, 1000, logits, nullptr, 0, 1024, 8, waves); }

  softmax_rows<<<(8192 * 32) / 256, 256, 0, stream>>>(logits, out);
}
